// AttentionCoeffGNN_multihead_fixed_2405181686018
// MI455X (gfx1250) — matrix operand order repaired and checked by reading, not compiled, not run on hardware
//
#include <hip/hip_runtime.h>

// Problem constants
#define B_    128
#define P_    256
#define H_    512
#define HEADS 8
#define HD    64
#define RANK  32
#define EPS_  1e-5f
#define ALPHA 0.9f
// logits scale: 1/(sqrt(P)*TAU) = 1/(16*0.3)
#define LSCL  (1.0f/(16.0f*0.3f))

typedef _Float16 v16h __attribute__((ext_vector_type(16)));
typedef _Float16 v8h  __attribute__((ext_vector_type(8)));
typedef float    v8f  __attribute__((ext_vector_type(8)));

// ---------------------------------------------------------------------------
// WMMA tile load/store helpers (wave32, V_WMMA_F32_16X16X32_F16 layouts)
// A: 16x32 f16.  lane: m=lane&15, half=lane>>4.
//   elements j -> K = 16*half + j
// B: 32x16 f16.  lane: n=lane&15, kh=lane>>4.  elements j -> K = 16*kh + j
// C/D: 16x16 f32. lane: n=lane&15, half=lane>>4.  reg r -> M = r + 8*half
// ---------------------------------------------------------------------------
static __device__ __forceinline__ v16h load_a_f16(const _Float16* tile, size_t lda, int lane) {
  int m = lane & 15, half = lane >> 4;
  const _Float16* p = tile + (size_t)m * lda + 16 * half;
  v8h lo = *(const v8h*)p;
  v8h hi = *(const v8h*)(p + 8);
  v16h a;
#pragma unroll
  for (int j = 0; j < 8; ++j) { a[j] = lo[j]; a[8 + j] = hi[j]; }
  return a;
}

static __device__ __forceinline__ v16h load_a_f32(const float* tile, size_t lda, int lane) {
  int m = lane & 15, half = lane >> 4;
  const float* p = tile + (size_t)m * lda + 16 * half;
  v8f lo = *(const v8f*)p;
  v8f hi = *(const v8f*)(p + 8);
  v16h a;
#pragma unroll
  for (int j = 0; j < 8; ++j) { a[j] = (_Float16)lo[j]; a[8 + j] = (_Float16)hi[j]; }
  return a;
}

static __device__ __forceinline__ v16h load_b_f32(const float* tile, size_t ldb, int lane) {
  int n = lane & 15, kh = lane >> 4;
  const float* p = tile + (size_t)n * ldb + 16 * kh;
  v8f q0 = *(const v8f*)p;
  v8f q1 = *(const v8f*)(p + 8);
  v16h b;
#pragma unroll
  for (int j = 0; j < 8; ++j) { b[j] = (_Float16)q0[j]; b[8 + j] = (_Float16)q1[j]; }
  return b;
}

static __device__ __forceinline__ v16h load_b_f16(const _Float16* tile, size_t ldb, int lane) {
  int n = lane & 15, kh = lane >> 4;
  const _Float16* p = tile + (size_t)n * ldb + 16 * kh;
  v8h q0 = *(const v8h*)p;
  v8h q1 = *(const v8h*)(p + 8);
  v16h b;
#pragma unroll
  for (int j = 0; j < 8; ++j) { b[j] = q0[j]; b[8 + j] = q1[j]; }
  return b;
}

static __device__ __forceinline__ void store_c_f32(float* out, size_t ldc, int lane,
                                                   v8f acc, float bias, bool relu) {
  int n = lane & 15, half = lane >> 4;
#pragma unroll
  for (int r = 0; r < 8; ++r) {
    float v = acc[r] + bias;
    if (relu) v = fmaxf(v, 0.f);
    out[(size_t)(r + 8 * half) * ldc + n] = v;
  }
}

static __device__ __forceinline__ void store_c_f16(_Float16* out, size_t ldc, int lane,
                                                   v8f acc, float bias, bool relu) {
  int n = lane & 15, half = lane >> 4;
#pragma unroll
  for (int r = 0; r < 8; ++r) {
    float v = acc[r] + bias;
    if (relu) v = fmaxf(v, 0.f);
    out[(size_t)(r + 8 * half) * ldc + n] = (_Float16)v;
  }
}

static __device__ __forceinline__ float wsum(float v) {
#pragma unroll
  for (int o = 16; o > 0; o >>= 1) v += __shfl_xor(v, o);
  return v;
}

// ---------------------------------------------------------------------------
// K0: per-head closed-form stats for the scalar-projection + LayerNorm algebra.
// st layout (floats): [0]A[8] [8]Bq[8] [16]Ck[8] [24]D[8]
//   [32]qVw [40]qC [48]qVb [56]kVw [64]kC [72]kVb [80]vVw [88]vC [96]vVb
//   [128..639] vhat_w[512]   [640..1151] vhat_b[512]
// ---------------------------------------------------------------------------
__global__ __launch_bounds__(256) void k_stats(const float* qw, const float* qb,
                                               const float* kw, const float* kb,
                                               const float* vw, const float* vb,
                                               float* st) {
  int head = threadIdx.x >> 5, lane = threadIdx.x & 31;
  int c0 = head * HD + lane, c1 = c0 + 32;
  float qw0 = qw[c0], qw1 = qw[c1], qb0 = qb[c0], qb1 = qb[c1];
  float kw0 = kw[c0], kw1 = kw[c1], kb0 = kb[c0], kb1 = kb[c1];
  float vw0 = vw[c0], vw1 = vw[c1], vb0 = vb[c0], vb1 = vb[c1];
  const float inv64 = 1.f / 64.f;
  float mwq = wsum(qw0 + qw1) * inv64, mbq = wsum(qb0 + qb1) * inv64;
  float mwk = wsum(kw0 + kw1) * inv64, mbk = wsum(kb0 + kb1) * inv64;
  float mwv = wsum(vw0 + vw1) * inv64, mbv = wsum(vb0 + vb1) * inv64;
  float cqw0 = qw0 - mwq, cqw1 = qw1 - mwq, cqb0 = qb0 - mbq, cqb1 = qb1 - mbq;
  float ckw0 = kw0 - mwk, ckw1 = kw1 - mwk, ckb0 = kb0 - mbk, ckb1 = kb1 - mbk;
  float cvw0 = vw0 - mwv, cvw1 = vw1 - mwv, cvb0 = vb0 - mbv, cvb1 = vb1 - mbv;
  float qVw = wsum(cqw0 * cqw0 + cqw1 * cqw1) * inv64;
  float qC  = wsum(cqw0 * cqb0 + cqw1 * cqb1) * inv64;
  float qVb = wsum(cqb0 * cqb0 + cqb1 * cqb1) * inv64;
  float kVw = wsum(ckw0 * ckw0 + ckw1 * ckw1) * inv64;
  float kC  = wsum(ckw0 * ckb0 + ckw1 * ckb1) * inv64;
  float kVb = wsum(ckb0 * ckb0 + ckb1 * ckb1) * inv64;
  float vVw = wsum(cvw0 * cvw0 + cvw1 * cvw1) * inv64;
  float vC  = wsum(cvw0 * cvb0 + cvw1 * cvb1) * inv64;
  float vVb = wsum(cvb0 * cvb0 + cvb1 * cvb1) * inv64;
  float A  = wsum(cqw0 * ckw0 + cqw1 * ckw1);
  float Bq = wsum(cqw0 * ckb0 + cqw1 * ckb1);
  float Ck = wsum(cqb0 * ckw0 + cqb1 * ckw1);
  float Dd = wsum(cqb0 * ckb0 + cqb1 * ckb1);
  if (lane == 0) {
    st[0 + head] = A;   st[8 + head] = Bq;  st[16 + head] = Ck; st[24 + head] = Dd;
    st[32 + head] = qVw; st[40 + head] = qC; st[48 + head] = qVb;
    st[56 + head] = kVw; st[64 + head] = kC; st[72 + head] = kVb;
    st[80 + head] = vVw; st[88 + head] = vC; st[96 + head] = vVb;
  }
  st[128 + c0] = cvw0; st[128 + c1] = cvw1;
  st[640 + c0] = cvb0; st[640 + c1] = cvb1;
}

// ---------------------------------------------------------------------------
// K1: per-position, per-head LN inverse scales for Q/K/V
// ---------------------------------------------------------------------------
__global__ __launch_bounds__(256) void k_scales(const float* x, const float* st,
                                                float* invq, float* invk, float* invv) {
  int idx = blockIdx.x * 256 + threadIdx.x;  // 0..B*P-1
  float xv = x[idx], x2 = xv * xv;
#pragma unroll
  for (int h = 0; h < 8; ++h) {
    invq[idx * 8 + h] = rsqrtf(x2 * st[32 + h] + 2.f * xv * st[40 + h] + st[48 + h] + EPS_);
    invk[idx * 8 + h] = rsqrtf(x2 * st[56 + h] + 2.f * xv * st[64 + h] + st[72 + h] + EPS_);
    invv[idx * 8 + h] = rsqrtf(x2 * st[80 + h] + 2.f * xv * st[88 + h] + st[96 + h] + EPS_);
  }
}

// ---------------------------------------------------------------------------
// Kres: res = x @ res_w.T + res_b   (128x256)@(256x512) via WMMA f16
// Two accumulators break the WMMA D->C dependency chain (hazard NOPs).
// ---------------------------------------------------------------------------
__global__ __launch_bounds__(256) void k_resg(const float* x, const float* resw,
                                              const float* resb, float* out) {
  int g = blockIdx.x * 8 + (threadIdx.x >> 5);
  int lane = threadIdx.x & 31;
  int mt = g & 7, nt = g >> 3;  // 8 x 32 tiles
  v8f acc0 = {}, acc1 = {};
#pragma unroll
  for (int kb = 0; kb < 256; kb += 64) {
    v16h a0 = load_a_f32(x + (size_t)(mt * 16) * 256 + kb, 256, lane);
    v16h b0 = load_b_f32(resw + (size_t)(nt * 16) * 256 + kb, 256, lane);
    acc0 = __builtin_amdgcn_wmma_f32_16x16x32_f16(false, a0, false, b0, (short)0, acc0, false, false);
    v16h a1 = load_a_f32(x + (size_t)(mt * 16) * 256 + kb + 32, 256, lane);
    v16h b1 = load_b_f32(resw + (size_t)(nt * 16) * 256 + kb + 32, 256, lane);
    acc1 = __builtin_amdgcn_wmma_f32_16x16x32_f16(false, a1, false, b1, (short)0, acc1, false, false);
  }
  v8f acc = acc0 + acc1;
  float bias = resb[nt * 16 + (lane & 15)];
  store_c_f32(out + (size_t)(mt * 16) * 512 + nt * 16, 512, lane, acc, bias, false);
}

// ---------------------------------------------------------------------------
// K2: attention in closed form. Block = (b, quarter of q). Thread layout
// pass1: 64 q x 4 k-slices, online per-head softmax stats + S0/S1.
// pass2: 256 k, loop q -> coalesced attn_mean writes.
// ---------------------------------------------------------------------------
__global__ __launch_bounds__(256) void k_attn(const float* x, const float* st,
                                              const float* invq, const float* invk,
                                              const float* invv,
                                              float* S0g, float* S1g, float* amean) {
  __shared__ float xs[256];
  __shared__ float rk[8][256];
  __shared__ float rv[8][256];
  __shared__ float e1s[8][64], e0s[8][64], ms[8][64], siv[8][64];
  int b = blockIdx.x >> 2, qc = blockIdx.x & 3;
  int tid = threadIdx.x;
  int pos = b * 256 + tid;
  xs[tid] = x[pos];
#pragma unroll
  for (int h = 0; h < 8; ++h) {
    rk[h][tid] = invk[pos * 8 + h];
    rv[h][tid] = invv[pos * 8 + h];
  }
  __syncthreads();

  float Ah[8], Bh[8], Ch[8], Dh[8];
#pragma unroll
  for (int h = 0; h < 8; ++h) { Ah[h] = st[h]; Bh[h] = st[8 + h]; Ch[h] = st[16 + h]; Dh[h] = st[24 + h]; }

  int ql = tid >> 2, ks = tid & 3;
  int q = qc * 64 + ql;
  float xq = xs[q];
  float e1[8], e0[8], m[8];
#pragma unroll
  for (int h = 0; h < 8; ++h) {
    float rq = invq[((size_t)b * 256 + q) * 8 + h];
    e1[h] = rq * (xq * Ah[h] + Ch[h]) * LSCL;
    e0[h] = rq * (xq * Bh[h] + Dh[h]) * LSCL;
    m[h] = -1e30f;
  }
  int k0 = ks * 64;
  for (int kk = 0; kk < 64; ++kk) {
    int k = k0 + kk;
    float xk = xs[k];
#pragma unroll
    for (int h = 0; h < 8; ++h) {
      float l = rk[h][k] * (xk * e1[h] + e0[h]);
      m[h] = fmaxf(m[h], l);
    }
  }
#pragma unroll
  for (int h = 0; h < 8; ++h) {
    m[h] = fmaxf(m[h], __shfl_xor(m[h], 1));
    m[h] = fmaxf(m[h], __shfl_xor(m[h], 2));
  }
  float s[8] = {}, s0[8] = {}, s1[8] = {};
  for (int kk = 0; kk < 64; ++kk) {
    int k = k0 + kk;
    float xk = xs[k];
#pragma unroll
    for (int h = 0; h < 8; ++h) {
      float l = rk[h][k] * (xk * e1[h] + e0[h]);
      float e = __expf(l - m[h]);
      s[h] += e;
      float w = e * rv[h][k];
      s0[h] += w;
      s1[h] += w * xk;
    }
  }
#pragma unroll
  for (int h = 0; h < 8; ++h) {
    s[h] += __shfl_xor(s[h], 1);  s[h] += __shfl_xor(s[h], 2);
    s0[h] += __shfl_xor(s0[h], 1); s0[h] += __shfl_xor(s0[h], 2);
    s1[h] += __shfl_xor(s1[h], 1); s1[h] += __shfl_xor(s1[h], 2);
  }
  if (ks == 0) {
#pragma unroll
    for (int h = 0; h < 8; ++h) {
      float si = 1.f / s[h];
      ms[h][ql] = m[h]; siv[h][ql] = si; e1s[h][ql] = e1[h]; e0s[h][ql] = e0[h];
      S0g[((size_t)b * 8 + h) * 256 + q] = s0[h] * si;
      S1g[((size_t)b * 8 + h) * 256 + q] = s1[h] * si;
    }
  }
  __syncthreads();

  // pass 2: thread = k column, loop q rows -> coalesced attn_mean
  int k = tid;
  float xk = xs[k];
  float rkl[8];
#pragma unroll
  for (int h = 0; h < 8; ++h) rkl[h] = rk[h][k];
  float* arow = amean + ((size_t)b * 256 + qc * 64) * 256 + k;
  for (int j = 0; j < 64; ++j) {
    float acc = 0.f;
#pragma unroll
    for (int h = 0; h < 8; ++h) {
      float l = rkl[h] * (xk * e1s[h][j] + e0s[h][j]);
      acc += __expf(l - ms[h][j]) * siv[h][j];
    }
    arow[(size_t)j * 256] = acc * 0.125f;
  }
}

// ---------------------------------------------------------------------------
// K3: h = vhat_w*S1 + vhat_b*S0 + alpha*res ; affine LN over 512 ch ; -> f16 z
// ---------------------------------------------------------------------------
__global__ __launch_bounds__(256) void k_combine(const float* st, const float* S0g,
                                                 const float* S1g, const float* res,
                                                 const float* lng, const float* lnb,
                                                 _Float16* z16) {
  __shared__ float red[16];
  __shared__ float mv[2];
  int b = blockIdx.x >> 8, p = blockIdx.x & 255;
  int tid = threadIdx.x;
  float v0, v1;
  {
    int c = tid, h = c >> 6;
    v0 = st[128 + c] * S1g[((size_t)b * 8 + h) * 256 + p] +
         st[640 + c] * S0g[((size_t)b * 8 + h) * 256 + p] +
         ALPHA * res[(size_t)b * 512 + c];
  }
  {
    int c = tid + 256, h = c >> 6;
    v1 = st[128 + c] * S1g[((size_t)b * 8 + h) * 256 + p] +
         st[640 + c] * S0g[((size_t)b * 8 + h) * 256 + p] +
         ALPHA * res[(size_t)b * 512 + c];
  }
  float s = v0 + v1, sq = v0 * v0 + v1 * v1;
#pragma unroll
  for (int o = 16; o > 0; o >>= 1) { s += __shfl_xor(s, o); sq += __shfl_xor(sq, o); }
  int wave = tid >> 5, lane = tid & 31;
  if (lane == 0) { red[wave] = s; red[8 + wave] = sq; }
  __syncthreads();
  if (tid == 0) {
    float ts = 0.f, tq = 0.f;
#pragma unroll
    for (int w = 0; w < 8; ++w) { ts += red[w]; tq += red[8 + w]; }
    float mean = ts * (1.f / 512.f);
    float var = tq * (1.f / 512.f) - mean * mean;
    mv[0] = mean; mv[1] = rsqrtf(var + EPS_);
  }
  __syncthreads();
  float mean = mv[0], inv = mv[1];
  size_t base = (size_t)b * 131072 + (size_t)p * 512;
  {
    int c = tid;
    z16[base + c] = (_Float16)((v0 - mean) * inv * lng[c] + lnb[c]);
  }
  {
    int c = tid + 256;
    z16[base + c] = (_Float16)((v1 - mean) * inv * lng[c] + lnb[c]);
  }
}

// ---------------------------------------------------------------------------
// K4: z(128x131072,f16) @ w0.T(131072x512,f32->f16) -> split-K partials.
// Grid: 32 N-tiles x 8 K-chunks; 8 waves/block each own 2048 of K.
// w0 streamed from HBM once via the CDNA5 async global->LDS path (ASYNCcnt),
// 4-slot LDS ring per wave, 2 stages in flight; z stays resident in L2.
// ---------------------------------------------------------------------------
__global__ __launch_bounds__(256) void k_gemm1(const _Float16* z16, const float* w0,
                                               float* part) {
  __shared__ float lbuf[8][4][512];  // 8 waves x 4 slots x 2KB = 64KB
  int nt = blockIdx.x & 31, kc = blockIdx.x >> 5;
  int wave = threadIdx.x >> 5, lane = threadIdx.x & 31;
  size_t kbase0 = (size_t)(kc * 8 + wave) * 2048;
  const float* w0base = w0 + (size_t)(nt * 16) * 131072;
  unsigned lds0 = (unsigned)(uintptr_t)(&lbuf[wave][0][0]);

  // stage one 16x32 f32 B-tile (K step ks_) into LDS slot (ks_&3):
  // 4 x async b128 per lane; t = i*32+lane -> row=t>>3, 4-float seg=t&7
  auto issueB = [&](int ks_) {
    size_t kb_ = kbase0 + (size_t)ks_ * 32;
    unsigned lslot = lds0 + ((unsigned)ks_ & 3u) * 2048u;
#pragma unroll
    for (int i = 0; i < 4; ++i) {
      int t = i * 32 + lane;
      int row = t >> 3, seg = t & 7;
      const float* g = w0base + (size_t)row * 131072 + kb_ + seg * 4;
      unsigned l = lslot + (unsigned)t * 16u;
      asm volatile("global_load_async_to_lds_b128 %0, %1, off"
                   :: "v"(l), "v"((unsigned long long)(uintptr_t)g)
                   : "memory");
    }
  };
  issueB(0);
  issueB(1);

  v8f acc[8] = {};
  int n = lane & 15, kh = lane >> 4;
  const float* lrd0 = &lbuf[wave][0][0];
  for (int ks = 0; ks < 64; ++ks) {
    // oldest stage is complete once at most the next stage (4 ops) is pending
    if (ks < 63) {
      asm volatile("s_wait_asynccnt %0" :: "n"(4) : "memory");
    } else {
      asm volatile("s_wait_asynccnt %0" :: "n"(0) : "memory");
    }
    const float* lp = lrd0 + (ks & 3) * 512 + n * 32 + 16 * kh;
    v8f q0 = *(const v8f*)lp;
    v8f q1 = *(const v8f*)(lp + 8);
    if (ks + 2 < 64) issueB(ks + 2);
    v16h b;
#pragma unroll
    for (int j = 0; j < 8; ++j) { b[j] = (_Float16)q0[j]; b[8 + j] = (_Float16)q1[j]; }
    size_t kb = kbase0 + (size_t)ks * 32;
#pragma unroll
    for (int mt = 0; mt < 8; ++mt) {
      v16h a = load_a_f16(z16 + (size_t)(mt * 16) * 131072 + kb, 131072, lane);
      acc[mt] = __builtin_amdgcn_wmma_f32_16x16x32_f16(false, a, false, b, (short)0, acc[mt], false, false);
    }
  }
  float* pout = part + (size_t)(kc * 8 + wave) * 65536;
#pragma unroll
  for (int mt = 0; mt < 8; ++mt)
    store_c_f32(pout + (size_t)(mt * 16) * 512 + nt * 16, 512, lane, acc[mt], 0.f, false);
}

// K5: deterministic split-K reduction + bias + relu -> z1 (f16)
__global__ __launch_bounds__(256) void k_reduce(const float* part, const float* bb0,
                                                _Float16* z1) {
  int idx = blockIdx.x * 256 + threadIdx.x;  // 0..65535
  float s = 0.f;
#pragma unroll 8
  for (int j = 0; j < 64; ++j) s += part[(size_t)j * 65536 + idx];
  s += bb0[idx & 511];
  z1[idx] = (_Float16)fmaxf(s, 0.f);
}

// K6: z1(128x512) @ w1.T(512x512) + bb1, relu -> z2 (f16), dual accumulators
__global__ __launch_bounds__(256) void k_gemm2(const _Float16* z1, const float* w1,
                                               const float* bb1, _Float16* z2) {
  int g = blockIdx.x * 8 + (threadIdx.x >> 5);
  int lane = threadIdx.x & 31;
  int mt = g & 7, nt = g >> 3;  // 8 x 32 tiles
  v8f acc0 = {}, acc1 = {};
#pragma unroll 2
  for (int kb = 0; kb < 512; kb += 64) {
    v16h a0 = load_a_f16(z1 + (size_t)(mt * 16) * 512 + kb, 512, lane);
    v16h b0 = load_b_f32(w1 + (size_t)(nt * 16) * 512 + kb, 512, lane);
    acc0 = __builtin_amdgcn_wmma_f32_16x16x32_f16(false, a0, false, b0, (short)0, acc0, false, false);
    v16h a1 = load_a_f16(z1 + (size_t)(mt * 16) * 512 + kb + 32, 512, lane);
    v16h b1 = load_b_f32(w1 + (size_t)(nt * 16) * 512 + kb + 32, 512, lane);
    acc1 = __builtin_amdgcn_wmma_f32_16x16x32_f16(false, a1, false, b1, (short)0, acc1, false, false);
  }
  v8f acc = acc0 + acc1;
  float bias = bb1[nt * 16 + (lane & 15)];
  store_c_f16(z2 + (size_t)(mt * 16) * 512 + nt * 16, 512, lane, acc, bias, true);
}

// K7: z2(128x512) @ w2.T(512x16384) + bb2 -> U|Vm (f16), dual accumulators
__global__ __launch_bounds__(256) void k_gemm3(const _Float16* z2, const float* w2,
                                               const float* bb2, _Float16* uv) {
  int g = blockIdx.x * 8 + (threadIdx.x >> 5);  // 0..8191
  int lane = threadIdx.x & 31;
  int mt = g & 7, nt = g >> 3;  // 8 x 1024 tiles
  v8f acc0 = {}, acc1 = {};
#pragma unroll 2
  for (int kb = 0; kb < 512; kb += 64) {
    v16h a0 = load_a_f16(z2 + (size_t)(mt * 16) * 512 + kb, 512, lane);
    v16h b0 = load_b_f32(w2 + (size_t)(nt * 16) * 512 + kb, 512, lane);
    acc0 = __builtin_amdgcn_wmma_f32_16x16x32_f16(false, a0, false, b0, (short)0, acc0, false, false);
    v16h a1 = load_a_f16(z2 + (size_t)(mt * 16) * 512 + kb + 32, 512, lane);
    v16h b1 = load_b_f32(w2 + (size_t)(nt * 16) * 512 + kb + 32, 512, lane);
    acc1 = __builtin_amdgcn_wmma_f32_16x16x32_f16(false, a1, false, b1, (short)0, acc1, false, false);
  }
  v8f acc = acc0 + acc1;
  float bias = bb2[nt * 16 + (lane & 15)];
  store_c_f16(uv + (size_t)(mt * 16) * 16384 + nt * 16, 16384, lane, acc, bias, false);
}

// K8: coeffs[b] = U[b](256x32) @ Vm[b].T(32x256) -- one WMMA per tile (K=32)
__global__ __launch_bounds__(256) void k_gemm4(const _Float16* uv, float* coeffs) {
  int g = blockIdx.x * 8 + (threadIdx.x >> 5);  // 0..32767
  int lane = threadIdx.x & 31;
  int b = g >> 8, t = g & 255;
  int mt = t & 15, nt = t >> 4;
  const _Float16* U = uv + (size_t)b * 16384;
  const _Float16* Vm = U + 8192;
  v16h a = load_a_f16(U + (size_t)(mt * 16) * 32, 32, lane);
  v16h bm = load_b_f16(Vm + (size_t)(nt * 16) * 32, 32, lane);
  v8f acc = {};
  acc = __builtin_amdgcn_wmma_f32_16x16x32_f16(false, a, false, bm, (short)0, acc, false, false);
  store_c_f32(coeffs + ((size_t)b * 256 + mt * 16) * 256 + nt * 16, 256, lane, acc, 0.f, false);
}

// ---------------------------------------------------------------------------
extern "C" void kernel_launch(void* const* d_in, const int* in_sizes, int n_in,
                              void* d_out, int out_size, void* d_ws, size_t ws_size,
                              hipStream_t stream) {
  (void)in_sizes; (void)n_in; (void)out_size; (void)ws_size;
  const float* x     = (const float*)d_in[0];
  const float* qw    = (const float*)d_in[1];
  const float* qb    = (const float*)d_in[2];
  const float* kw    = (const float*)d_in[3];
  const float* kb    = (const float*)d_in[4];
  const float* vw    = (const float*)d_in[5];
  const float* vb    = (const float*)d_in[6];
  const float* res_w = (const float*)d_in[7];
  const float* res_b = (const float*)d_in[8];
  const float* ln_g  = (const float*)d_in[9];
  const float* ln_b  = (const float*)d_in[10];
  const float* w0    = (const float*)d_in[11];
  const float* bb0   = (const float*)d_in[12];
  const float* w1    = (const float*)d_in[13];
  const float* bb1   = (const float*)d_in[14];
  const float* w2    = (const float*)d_in[15];
  const float* bb2   = (const float*)d_in[16];

  char* wsb = (char*)d_ws;
  float*    st   = (float*)(wsb + 0);          // 8 KB stats
  float*    invq = (float*)(wsb + 8192);       // 1 MB
  float*    invk = (float*)(wsb + 1056768);    // 1 MB
  float*    invv = (float*)(wsb + 2105344);    // 1 MB
  float*    res  = (float*)(wsb + 3153920);    // 256 KB
  float*    S0g  = (float*)(wsb + 3416064);    // 1 MB
  float*    S1g  = (float*)(wsb + 4464640);    // 1 MB
  _Float16* z16  = (_Float16*)(wsb + 5513216); // 32 MB (B x P*H f16)
  float*    part = (float*)(wsb + 39067648);   // 16 MB split-K partials
  _Float16* z1   = (_Float16*)(wsb + 55844864);
  _Float16* z2   = (_Float16*)(wsb + 55975936);
  _Float16* uv   = (_Float16*)(wsb + 56107008); // 4 MB U|Vm f16

  float* coeffs = (float*)d_out;
  float* amean  = coeffs + (size_t)B_ * P_ * P_;

  k_stats  <<<1,     256, 0, stream>>>(qw, qb, kw, kb, vw, vb, st);
  k_scales <<<128,   256, 0, stream>>>(x, st, invq, invk, invv);
  k_resg   <<<32,    256, 0, stream>>>(x, res_w, res_b, res);
  k_attn   <<<512,   256, 0, stream>>>(x, st, invq, invk, invv, S0g, S1g, amean);
  k_combine<<<32768, 256, 0, stream>>>(st, S0g, S1g, res, ln_g, ln_b, z16);
  k_gemm1  <<<256,   256, 0, stream>>>(z16, w0, part);
  k_reduce <<<256,   256, 0, stream>>>(part, bb0, z1);
  k_gemm2  <<<32,    256, 0, stream>>>(z1, w1, bb1, z2);
  k_gemm3  <<<1024,  256, 0, stream>>>(z2, w2, bb2, uv);
  k_gemm4  <<<4096,  256, 0, stream>>>(uv, coeffs);
}
